// GATTNetConvHybrid_45191645888902
// MI455X (gfx1250) — matrix operand order repaired and checked by reading, not compiled, not run on hardware
//
#include <hip/hip_runtime.h>

#define NNODES 30000
#define NEDGES 480000
#define EPSBN  1e-5f

typedef __attribute__((ext_vector_type(16))) _Float16 v16h;
typedef __attribute__((ext_vector_type(8)))  float    v8f;

union V16H { v16h v; unsigned u[8]; };

__device__ __forceinline__ v8f wmma_f16f32(v16h a, v16h b, v8f c) {
  // D = A(16x32 f16) * B(32x16 f16) + C(16x16 f32)
  return __builtin_amdgcn_wmma_f32_16x16x32_f16(false, a, false, b, (short)0, c, false, false);
}

// order-preserving float <-> uint mapping for atomicMax on floats
__device__ __forceinline__ unsigned f32key(float f) {
  unsigned u = __float_as_uint(f);
  return (u & 0x80000000u) ? ~u : (u | 0x80000000u);
}
__device__ __forceinline__ float keyf32(unsigned u) {
  return __uint_as_float((u & 0x80000000u) ? (u & 0x7fffffffu) : ~u);
}

__device__ __forceinline__ float lrelu(float x, float a) { return x > 0.f ? x : a * x; }
__device__ __forceinline__ float eluf(float x) { return x > 0.f ? x : expm1f(x); }

// A-operand half-index for 16x16x32 f16 WMMA: pairs (K,K+1) contiguous
__device__ __forceinline__ int aoff(int kc, int p, int hi) {
  return (kc << 5) + 2 * p + ((p >= 4) ? 8 : 0) + hi * 8;
}

// ---------------------------------------------------------------------------
// Elementwise helpers
// ---------------------------------------------------------------------------
__global__ void cast_f32_f16_kernel(const float* __restrict__ x, _Float16* __restrict__ y, int n) {
  int t = blockIdx.x * blockDim.x + threadIdx.x;
  if (t < n) y[t] = (_Float16)x[t];
}

__global__ void count_kernel(const int* __restrict__ dst, int E, int* __restrict__ cnt) {
  int t = blockIdx.x * blockDim.x + threadIdx.x;
  if (t < E) atomicAdd(&cnt[dst[t]], 1);
}

// x1 = leaky_relu(acc, 0.01) -> xc16[:, 0:128] (f16)
__global__ void post_ec1_kernel(const float* __restrict__ acc, _Float16* __restrict__ xc16, int n) {
  int t = blockIdx.x * blockDim.x + threadIdx.x;
  if (t >= n * 128) return;
  int node = t >> 7, c = t & 127;
  float v = lrelu(acc[t], 0.01f);
  xc16[(size_t)node * 256 + c] = (_Float16)v;
}

// x2 = leaky_relu(acc/max(cnt,1), 0.01) -> xc16[:, 128:256] (f16)
__global__ void post_ec2_kernel(const float* __restrict__ acc, const int* __restrict__ cnt,
                                _Float16* __restrict__ xc16, int n) {
  int t = blockIdx.x * blockDim.x + threadIdx.x;
  if (t >= n * 128) return;
  int node = t >> 7, c = t & 127;
  float d = (float)(cnt[node] > 1 ? cnt[node] : 1);
  float v = lrelu(acc[t] / d, 0.01f);
  xc16[(size_t)node * 256 + 128 + c] = (_Float16)v;
}

// ---------------------------------------------------------------------------
// Weight prepack: pack f32 row-major W (Kdim x nReal) into the WMMA B-operand
// lane layout: element e of lane L holds W[kc*32 + e + 8*(L>=16) + 8*(e>=8)][nt*16 + L%16]
// ---------------------------------------------------------------------------
__global__ void prepack_b_kernel(_Float16* __restrict__ Pk, const float* __restrict__ W,
                                 int ldw, int nReal, int ntTiles, int kTiles,
                                 const float* __restrict__ bng, const float* __restrict__ bnv) {
  int t = blockIdx.x * blockDim.x + threadIdx.x;
  int total = kTiles * ntTiles * 512;
  if (t >= total) return;
  int e = t & 15, lane = (t >> 4) & 31, rest = t >> 9;
  int nt = rest % ntTiles, kc = rest / ntTiles;
  int K = kc * 32 + e + ((lane & 16) ? 8 : 0) + ((e >= 8) ? 8 : 0);
  int N = nt * 16 + (lane & 15);
  float v = 0.f;
  if (N < nReal) {
    v = W[(size_t)K * ldw + N];
    if (bng) v *= bng[N] * rsqrtf(bnv[N] + EPSBN);
  }
  Pk[t] = (_Float16)v;
}

// EdgeConv fc1+short factorization: virtual 128x512 weight [PA|PB|SA|SB]
//   PA = (W1a - W1b)*s1,  PB = W1b*s1,  SA = Wsa - Wsb,  SB = Wsb
__global__ void prepack_ec_fc1_kernel(_Float16* __restrict__ Pk, const float* __restrict__ W1,
                                      const float* __restrict__ Ws,
                                      const float* __restrict__ bng, const float* __restrict__ bnv) {
  int t = blockIdx.x * blockDim.x + threadIdx.x;
  if (t >= 4 * 32 * 512) return;  // kTiles=4, ntTiles=32
  int e = t & 15, lane = (t >> 4) & 31, rest = t >> 9;
  int nt = rest % 32, kc = rest / 32;
  int K = kc * 32 + e + ((lane & 16) ? 8 : 0) + ((e >= 8) ? 8 : 0);
  int N = nt * 16 + (lane & 15);
  float v;
  if (N < 128) {
    float s = bng[N] * rsqrtf(bnv[N] + EPSBN);
    v = (W1[(size_t)K * 128 + N] - W1[(size_t)(K + 128) * 128 + N]) * s;
  } else if (N < 256) {
    int j = N - 128;
    float s = bng[j] * rsqrtf(bnv[j] + EPSBN);
    v = W1[(size_t)(K + 128) * 128 + j] * s;
  } else if (N < 384) {
    int j = N - 256;
    v = Ws[(size_t)K * 128 + j] - Ws[(size_t)(K + 128) * 128 + j];
  } else {
    int j = N - 384;
    v = Ws[(size_t)(K + 128) * 128 + j];
  }
  Pk[t] = (_Float16)v;
}

// Fold BN shifts + biases: out[0:128)=bn1(fc1 bias), [128:256)=bn2(fc2 bias), [256:384)=b3+bshort
__global__ void bias_fold_kernel(const float* b1, const float* g1, const float* bb1,
                                 const float* m1, const float* v1,
                                 const float* b2, const float* g2, const float* bb2,
                                 const float* m2, const float* v2,
                                 const float* b3, const float* bs, float* __restrict__ out) {
  int c = threadIdx.x;
  if (c >= 128) return;
  float s1 = g1[c] * rsqrtf(v1[c] + EPSBN);
  out[c] = b1[c] * s1 + bb1[c] - m1[c] * s1;
  float s2 = g2[c] * rsqrtf(v2[c] + EPSBN);
  out[128 + c] = b2[c] * s2 + bb2[c] - m2[c] * s2;
  out[256 + c] = b3[c] + bs[c];
}

// ---------------------------------------------------------------------------
// Generic node-level WMMA GEMM: C[rowTiles*16 x nStore] f32 =
//   A(f16, row-major, lda) x Bp(prepacked) (+ bias)
// One wave per 16-row tile, 4 accumulator N-tiles per group.
// Explicit double-buffering of A and B registers across the K loop so the
// scheduler can overlap global loads with the WMMA chain (partial loadcnt
// waits instead of full drains). Requires kTiles even, ntTiles % 4 == 0.
// ---------------------------------------------------------------------------
__global__ __launch_bounds__(256) void gemm16_kernel(
    const _Float16* __restrict__ A, int lda, int Kdim,
    const _Float16* __restrict__ Bp, int ntTiles,
    float* __restrict__ C, int ldc, int nStore,
    const float* __restrict__ bias, int nRowTiles) {
  int wave = threadIdx.x >> 5, lane = threadIdx.x & 31;
  int tile = blockIdx.x * (blockDim.x >> 5) + wave;
  if (tile >= nRowTiles) return;
  int m = lane & 15, hi = (lane >> 4) & 1;
  const _Float16* Arow = A + (size_t)(tile * 16 + m) * lda;
  const v16h* B32 = (const v16h*)Bp;
  int kTiles = Kdim >> 5;
  for (int ntg = 0; ntg < ntTiles; ntg += 4) {
    v8f acc[4];
#pragma unroll
    for (int j = 0; j < 4; ++j)
#pragma unroll
      for (int i = 0; i < 8; ++i) acc[j][i] = 0.f;

    V16H a0, a1;
    v16h bA[4], bB[4];
#pragma unroll
    for (int p = 0; p < 8; ++p) a0.u[p] = *(const unsigned*)(Arow + aoff(0, p, hi));
#pragma unroll
    for (int j = 0; j < 4; ++j) bA[j] = B32[((size_t)(0 * ntTiles + ntg + j) << 5) + lane];

    for (int kc = 0; kc < kTiles; kc += 2) {
      // prefetch chunk kc+1 into the idle buffers
#pragma unroll
      for (int p = 0; p < 8; ++p) a1.u[p] = *(const unsigned*)(Arow + aoff(kc + 1, p, hi));
#pragma unroll
      for (int j = 0; j < 4; ++j) bB[j] = B32[((size_t)((kc + 1) * ntTiles + ntg + j) << 5) + lane];
#pragma unroll
      for (int j = 0; j < 4; ++j) acc[j] = wmma_f16f32(a0.v, bA[j], acc[j]);
      // prefetch chunk kc+2
      if (kc + 2 < kTiles) {
#pragma unroll
        for (int p = 0; p < 8; ++p) a0.u[p] = *(const unsigned*)(Arow + aoff(kc + 2, p, hi));
#pragma unroll
        for (int j = 0; j < 4; ++j) bA[j] = B32[((size_t)((kc + 2) * ntTiles + ntg + j) << 5) + lane];
      }
#pragma unroll
      for (int j = 0; j < 4; ++j) acc[j] = wmma_f16f32(a1.v, bB[j], acc[j]);
    }

#pragma unroll
    for (int j = 0; j < 4; ++j) {
      int col = (ntg + j) * 16 + m;
      if (col < nStore) {
        float bv = bias ? bias[col] : 0.f;
#pragma unroll
        for (int r = 0; r < 8; ++r) {
          int row = tile * 16 + r + hi * 8;
          C[(size_t)row * ldc + col] = acc[j][r] + bv;
        }
      }
    }
  }
}

// ---------------------------------------------------------------------------
// Per-edge ResMLP core: h1 = relu(PA[dst]+PB[src]+bias1)   (from node table PPS)
//                       h2 = relu(h1 @ W2' + bias2)        (WMMA, LDS restage)
//                       out= h2 @ W3 + (b3+bs) + SA[dst]+SB[src]
//                       atomic segment-sum into outAcc[dst]
// 8 waves/block, one 16-edge tile per wave; 32KB LDS staging.
// The two 128x128 GEMMs run as an 8-segment pipeline (segment = 4 N-tiles of
// one K-chunk) with alternating A/B register buffers for load/WMMA overlap.
// ---------------------------------------------------------------------------
__global__ __launch_bounds__(256) void edge_mlp_kernel(
    const int* __restrict__ src, const int* __restrict__ dst, int E,
    const float* __restrict__ PPS,                    // N x 512: PA|PB|SA|SB
    const _Float16* __restrict__ W2p, const _Float16* __restrict__ W3p,
    const float* __restrict__ biasFold,               // 384 floats
    float* __restrict__ outAcc) {                     // N x 128
  __shared__ _Float16 stage[8][16][128];
  int wave = threadIdx.x >> 5, lane = threadIdx.x & 31;
  int m = lane & 15, hi = (lane >> 4) & 1;
  int e0 = (blockIdx.x * 8 + wave) * 16;

  // prefetch next tile's edge indices into cache (global_prefetch_b8)
  if (lane == 0 && e0 + 128 < E) {
    __builtin_prefetch(src + e0 + 128, 0, 0);
    __builtin_prefetch(dst + e0 + 128, 0, 0);
  }

  // gather + fc1(bn1-folded) + relu -> LDS (row-major A tile)
  for (int idx = lane; idx < 16 * 128; idx += 32) {
    int r = idx >> 7, c = idx & 127;
    int e = e0 + r; if (e >= E) e = E - 1;
    int d = dst[e], s = src[e];
    float v = PPS[(size_t)d * 512 + c] + PPS[(size_t)s * 512 + 128 + c] + biasFold[c];
    stage[wave][r][c] = (_Float16)fmaxf(v, 0.f);
  }
  __syncthreads();

  v8f acc[8];
  auto zacc = [&]() {
#pragma unroll
    for (int nt = 0; nt < 8; ++nt)
#pragma unroll
      for (int i = 0; i < 8; ++i) acc[nt][i] = 0.f;
  };
  auto loadA = [&](V16H& a, int kc) {
#pragma unroll
    for (int p = 0; p < 8; ++p)
      a.u[p] = *(const unsigned*)&stage[wave][m][aoff(kc, p, hi)];
  };
  auto loadB4 = [&](v16h* b, const v16h* Bm, int kc, int g) {
#pragma unroll
    for (int j = 0; j < 4; ++j) b[j] = Bm[(kc * 8 + g + j) * 32 + lane];
  };
  auto fma4 = [&](int g, const V16H& a, const v16h* b) {
#pragma unroll
    for (int j = 0; j < 4; ++j) acc[g + j] = wmma_f16f32(a.v, b[j], acc[g + j]);
  };
  // 128x128 GEMM, pipelined over 8 segments (kc 0..3 x half 0/1)
  auto gemm128 = [&](const v16h* Bm) {
    V16H aA, aB;
    v16h b0[4], b1[4];
    loadA(aA, 0);
    loadB4(b0, Bm, 0, 0);
    loadB4(b1, Bm, 0, 4); loadA(aB, 1); fma4(0, aA, b0);   // s0 (kc0, aA)
    loadB4(b0, Bm, 1, 0);               fma4(4, aA, b1);   // s1
    loadB4(b1, Bm, 1, 4); loadA(aA, 2); fma4(0, aB, b0);   // s2 (kc1, aB)
    loadB4(b0, Bm, 2, 0);               fma4(4, aB, b1);   // s3
    loadB4(b1, Bm, 2, 4); loadA(aB, 3); fma4(0, aA, b0);   // s4 (kc2, aA)
    loadB4(b0, Bm, 3, 0);               fma4(4, aA, b1);   // s5
    loadB4(b1, Bm, 3, 4);               fma4(0, aB, b0);   // s6 (kc3, aB)
                                        fma4(4, aB, b1);   // s7
  };

  // GEMM1: h1 x W2'
  zacc();
  gemm128((const v16h*)W2p);
  __syncthreads();

  // relu(+bias2), restage as row-major f16
#pragma unroll
  for (int nt = 0; nt < 8; ++nt) {
    int col = nt * 16 + m;
    float bv = biasFold[128 + col];
#pragma unroll
    for (int r = 0; r < 8; ++r)
      stage[wave][r + hi * 8][col] = (_Float16)fmaxf(acc[nt][r] + bv, 0.f);
  }
  __syncthreads();

  // GEMM2: h2 x W3
  zacc();
  gemm128((const v16h*)W3p);

  // + residual (SA[dst]+SB[src]+b3+bs), atomic scatter into dst rows
#pragma unroll
  for (int nt = 0; nt < 8; ++nt) {
    int col = nt * 16 + m;
    float b3v = biasFold[256 + col];
#pragma unroll
    for (int r = 0; r < 8; ++r) {
      int row = r + hi * 8;
      int e = e0 + row;
      if (e < E) {
        int d = dst[e], s = src[e];
        float v = acc[nt][r] + b3v + PPS[(size_t)d * 512 + 256 + col] + PPS[(size_t)s * 512 + 384 + col];
        atomicAdd(&outAcc[(size_t)d * 128 + col], v);
      }
    }
  }
}

// ---------------------------------------------------------------------------
// GAT kernels (heads=4, per-head dim 128 for both layers)
// ---------------------------------------------------------------------------
__global__ void gat_scores_kernel(const float* __restrict__ H,
                                  const float* __restrict__ aS, const float* __restrict__ aD,
                                  float* __restrict__ asrc, float* __restrict__ adst,
                                  unsigned* __restrict__ emax, int n) {
  int t = blockIdx.x * blockDim.x + threadIdx.x;
  if (t >= n * 4) return;
  int node = t >> 2, h = t & 3;
  const float* Hp = H + (size_t)node * 512 + h * 128;
  const float* as_ = aS + h * 128;
  const float* ad_ = aD + h * 128;
  float sa = 0.f, sd = 0.f;
  for (int c = 0; c < 128; ++c) { float v = Hp[c]; sa += v * as_[c]; sd += v * ad_[c]; }
  asrc[t] = sa;
  adst[t] = sd;
  emax[t] = f32key(lrelu(sa + sd, 0.2f));  // seed segment-max with the self-loop score
}

__global__ void gat_max_kernel(const int* __restrict__ src, const int* __restrict__ dst, int E,
                               const float* __restrict__ asrc, const float* __restrict__ adst,
                               unsigned* __restrict__ emax) {
  int t = blockIdx.x * blockDim.x + threadIdx.x;
  if (t >= E * 4) return;
  int e = t >> 2, h = t & 3;
  int s = src[e], d = dst[e];
  float v = lrelu(asrc[s * 4 + h] + adst[d * 4 + h], 0.2f);
  atomicMax(&emax[d * 4 + h], f32key(v));
}

// one wave per (edge-or-selfloop, head)
__global__ __launch_bounds__(256) void gat_accum_kernel(
    const int* __restrict__ src, const int* __restrict__ dst, int E, int n,
    const float* __restrict__ H, const float* __restrict__ asrc, const float* __restrict__ adst,
    const unsigned* __restrict__ emax, float* __restrict__ denom, float* __restrict__ MSG) {
  int wave = threadIdx.x >> 5, lane = threadIdx.x & 31;
  long wid = (long)blockIdx.x * 8 + wave;
  if (wid >= (long)(E + n) * 4) return;
  int e = (int)(wid >> 2), h = (int)(wid & 3);
  int s, d;
  if (e < E) { s = src[e]; d = dst[e]; } else { s = e - E; d = s; }
  float ev = lrelu(asrc[s * 4 + h] + adst[d * 4 + h], 0.2f);
  float w = expf(ev - keyf32(emax[d * 4 + h]));
  if (lane == 0) atomicAdd(&denom[d * 4 + h], w);
  const float* hs = H + (size_t)s * 512 + h * 128;
  float* mg = MSG + (size_t)d * 512 + h * 128;
  for (int c = lane; c < 128; c += 32) atomicAdd(&mg[c], w * hs[c]);
}

__global__ void gat_final1_kernel(const float* __restrict__ MSG, const float* __restrict__ denom,
                                  const float* __restrict__ bias, _Float16* __restrict__ g116, int n) {
  int t = blockIdx.x * blockDim.x + threadIdx.x;
  if (t >= n * 512) return;
  int node = t >> 9, j = t & 511, h = j >> 7;
  float v = MSG[t] / denom[node * 4 + h] + bias[j];
  g116[t] = (_Float16)eluf(v);
}

__global__ void gat_final2_kernel(const float* __restrict__ MSG, const float* __restrict__ denom,
                                  const float* __restrict__ bias, _Float16* __restrict__ g216, int n) {
  int t = blockIdx.x * blockDim.x + threadIdx.x;
  if (t >= n * 128) return;
  int node = t >> 7, c = t & 127;
  float v = 0.f;
#pragma unroll
  for (int h = 0; h < 4; ++h) v += MSG[(size_t)node * 512 + h * 128 + c] / denom[node * 4 + h];
  v = 0.25f * v + bias[c];
  g216[t] = (_Float16)eluf(v);
}

// ---------------------------------------------------------------------------
static inline int cdiv(long a, long b) { return (int)((a + b - 1) / b); }

extern "C" void kernel_launch(void* const* d_in, const int* in_sizes, int n_in,
                              void* d_out, int out_size, void* d_ws, size_t ws_size,
                              hipStream_t stream) {
  (void)in_sizes; (void)n_in; (void)out_size; (void)ws_size;
  const int N = NNODES, E = NEDGES;

  const float* x = (const float*)d_in[0];
  const int* ei = (const int*)d_in[1];
  const int* src = ei;
  const int* dst = ei + E;
  // ec1: 2..17  ec2: 18..33  gat1: 34..37  gat2: 38..41  cls: 42..43
  const float *e1W1=(const float*)d_in[2],  *e1b1=(const float*)d_in[3],
              *e1g1=(const float*)d_in[4],  *e1bb1=(const float*)d_in[5],
              *e1m1=(const float*)d_in[6],  *e1v1=(const float*)d_in[7],
              *e1W2=(const float*)d_in[8],  *e1b2=(const float*)d_in[9],
              *e1g2=(const float*)d_in[10], *e1bb2=(const float*)d_in[11],
              *e1m2=(const float*)d_in[12], *e1v2=(const float*)d_in[13],
              *e1W3=(const float*)d_in[14], *e1b3=(const float*)d_in[15],
              *e1Ws=(const float*)d_in[16], *e1bs=(const float*)d_in[17];
  const float *e2W1=(const float*)d_in[18], *e2b1=(const float*)d_in[19],
              *e2g1=(const float*)d_in[20], *e2bb1=(const float*)d_in[21],
              *e2m1=(const float*)d_in[22], *e2v1=(const float*)d_in[23],
              *e2W2=(const float*)d_in[24], *e2b2=(const float*)d_in[25],
              *e2g2=(const float*)d_in[26], *e2bb2=(const float*)d_in[27],
              *e2m2=(const float*)d_in[28], *e2v2=(const float*)d_in[29],
              *e2W3=(const float*)d_in[30], *e2b3=(const float*)d_in[31],
              *e2Ws=(const float*)d_in[32], *e2bs=(const float*)d_in[33];
  const float *g1W=(const float*)d_in[34], *g1aS=(const float*)d_in[35],
              *g1aD=(const float*)d_in[36], *g1b=(const float*)d_in[37];
  const float *g2W=(const float*)d_in[38], *g2aS=(const float*)d_in[39],
              *g2aD=(const float*)d_in[40], *g2b=(const float*)d_in[41];
  const float *clsW=(const float*)d_in[42], *clsB=(const float*)d_in[43];
  float* out = (float*)d_out;

  // workspace carve-out (≈ 202 MB)
  char* W = (char*)d_ws;
  size_t cur = 0;
  auto take = [&](size_t b) { size_t o = cur; cur += (b + 255) & ~(size_t)255; return o; };
  _Float16* X16  = (_Float16*)(W + take((size_t)N * 128 * 2));
  _Float16* XC16 = (_Float16*)(W + take((size_t)N * 256 * 2));
  float*    PPS  = (float*)(W + take((size_t)N * 512 * 4));   // PA|PB|SA|SB, later H
  float*    MSG  = (float*)(W + take((size_t)N * 512 * 4));
  float*    ACC  = (float*)(W + take((size_t)N * 128 * 4));
  _Float16* G116 = (_Float16*)(W + take((size_t)N * 512 * 2));
  _Float16* G216 = (_Float16*)(W + take((size_t)N * 128 * 2));
  int*      CNT  = (int*)(W + take((size_t)N * 4));
  float*    ASRC = (float*)(W + take((size_t)N * 4 * 4));
  float*    ADST = (float*)(W + take((size_t)N * 4 * 4));
  unsigned* EMAX = (unsigned*)(W + take((size_t)N * 4 * 4));
  float*    DEN  = (float*)(W + take((size_t)N * 4 * 4));
  _Float16* PkE1  = (_Float16*)(W + take((size_t)128 * 512 * 2));
  _Float16* PkE1W2= (_Float16*)(W + take((size_t)128 * 128 * 2));
  _Float16* PkE1W3= (_Float16*)(W + take((size_t)128 * 128 * 2));
  _Float16* PkE2  = (_Float16*)(W + take((size_t)128 * 512 * 2));
  _Float16* PkE2W2= (_Float16*)(W + take((size_t)128 * 128 * 2));
  _Float16* PkE2W3= (_Float16*)(W + take((size_t)128 * 128 * 2));
  _Float16* PkG1  = (_Float16*)(W + take((size_t)256 * 512 * 2));
  _Float16* PkG2  = (_Float16*)(W + take((size_t)512 * 512 * 2));
  _Float16* PkCLS = (_Float16*)(W + take((size_t)128 * 128 * 2));  // padded to 128 cols
  float*    BF1   = (float*)(W + take(384 * 4));
  float*    BF2   = (float*)(W + take(384 * 4));

  const int rowTiles = N / 16;  // 1875
  dim3 b256(256);

  // --- weight prepacks + bias folds ---
  prepack_ec_fc1_kernel<<<cdiv(65536,256), b256, 0, stream>>>(PkE1, e1W1, e1Ws, e1g1, e1v1);
  prepack_b_kernel<<<cdiv(16384,256), b256, 0, stream>>>(PkE1W2, e1W2, 128, 128, 8, 4, e1g2, e1v2);
  prepack_b_kernel<<<cdiv(16384,256), b256, 0, stream>>>(PkE1W3, e1W3, 128, 128, 8, 4, nullptr, nullptr);
  prepack_ec_fc1_kernel<<<cdiv(65536,256), b256, 0, stream>>>(PkE2, e2W1, e2Ws, e2g1, e2v1);
  prepack_b_kernel<<<cdiv(16384,256), b256, 0, stream>>>(PkE2W2, e2W2, 128, 128, 8, 4, e2g2, e2v2);
  prepack_b_kernel<<<cdiv(16384,256), b256, 0, stream>>>(PkE2W3, e2W3, 128, 128, 8, 4, nullptr, nullptr);
  prepack_b_kernel<<<cdiv(131072,256), b256, 0, stream>>>(PkG1, g1W, 512, 512, 32, 8, nullptr, nullptr);
  prepack_b_kernel<<<cdiv(262144,256), b256, 0, stream>>>(PkG2, g2W, 512, 512, 32, 16, nullptr, nullptr);
  prepack_b_kernel<<<cdiv(16384,256), b256, 0, stream>>>(PkCLS, clsW, 40, 40, 8, 4, nullptr, nullptr);
  bias_fold_kernel<<<1, 128, 0, stream>>>(e1b1,e1g1,e1bb1,e1m1,e1v1, e1b2,e1g2,e1bb2,e1m2,e1v2, e1b3,e1bs, BF1);
  bias_fold_kernel<<<1, 128, 0, stream>>>(e2b1,e2g1,e2bb1,e2m1,e2v1, e2b2,e2g2,e2bb2,e2m2,e2v2, e2b3,e2bs, BF2);

  hipMemsetAsync(CNT, 0, (size_t)N * 4, stream);
  count_kernel<<<cdiv(E,256), b256, 0, stream>>>(dst, E, CNT);

  // --- EdgeConv 1 ---
  cast_f32_f16_kernel<<<cdiv((long)N*128,256), b256, 0, stream>>>(x, X16, N * 128);
  gemm16_kernel<<<cdiv(rowTiles,8), b256, 0, stream>>>(X16, 128, 128, PkE1, 32, PPS, 512, 512, nullptr, rowTiles);
  hipMemsetAsync(ACC, 0, (size_t)N * 128 * 4, stream);
  edge_mlp_kernel<<<cdiv(E,128), b256, 0, stream>>>(src, dst, E, PPS, PkE1W2, PkE1W3, BF1, ACC);
  post_ec1_kernel<<<cdiv((long)N*128,256), b256, 0, stream>>>(ACC, XC16, N);

  // --- EdgeConv 2 (mean aggregation) ---
  gemm16_kernel<<<cdiv(rowTiles,8), b256, 0, stream>>>(XC16, 256, 128, PkE2, 32, PPS, 512, 512, nullptr, rowTiles);
  hipMemsetAsync(ACC, 0, (size_t)N * 128 * 4, stream);
  edge_mlp_kernel<<<cdiv(E,128), b256, 0, stream>>>(src, dst, E, PPS, PkE2W2, PkE2W3, BF2, ACC);
  post_ec2_kernel<<<cdiv((long)N*128,256), b256, 0, stream>>>(ACC, CNT, XC16, N);

  // --- GAT 1 (concat heads) ---
  gemm16_kernel<<<cdiv(rowTiles,8), b256, 0, stream>>>(XC16, 256, 256, PkG1, 32, PPS, 512, 512, nullptr, rowTiles);
  gat_scores_kernel<<<cdiv((long)N*4,256), b256, 0, stream>>>(PPS, g1aS, g1aD, ASRC, ADST, EMAX, N);
  gat_max_kernel<<<cdiv((long)E*4,256), b256, 0, stream>>>(src, dst, E, ASRC, ADST, EMAX);
  hipMemsetAsync(MSG, 0, (size_t)N * 512 * 4, stream);
  hipMemsetAsync(DEN, 0, (size_t)N * 4 * 4, stream);
  gat_accum_kernel<<<cdiv((long)(E+N)*4,8), b256, 0, stream>>>(src, dst, E, N, PPS, ASRC, ADST, EMAX, DEN, MSG);
  gat_final1_kernel<<<cdiv((long)N*512,256), b256, 0, stream>>>(MSG, DEN, g1b, G116, N);

  // --- GAT 2 (mean heads) ---
  gemm16_kernel<<<cdiv(rowTiles,8), b256, 0, stream>>>(G116, 512, 512, PkG2, 32, PPS, 512, 512, nullptr, rowTiles);
  gat_scores_kernel<<<cdiv((long)N*4,256), b256, 0, stream>>>(PPS, g2aS, g2aD, ASRC, ADST, EMAX, N);
  gat_max_kernel<<<cdiv((long)E*4,256), b256, 0, stream>>>(src, dst, E, ASRC, ADST, EMAX);
  hipMemsetAsync(MSG, 0, (size_t)N * 512 * 4, stream);
  hipMemsetAsync(DEN, 0, (size_t)N * 4 * 4, stream);
  gat_accum_kernel<<<cdiv((long)(E+N)*4,8), b256, 0, stream>>>(src, dst, E, N, PPS, ASRC, ADST, EMAX, DEN, MSG);
  gat_final2_kernel<<<cdiv((long)N*128,256), b256, 0, stream>>>(MSG, DEN, g2b, G216, N);

  // --- classifier (output padded to 8 N-tiles, stores bounded to 40 cols) ---
  gemm16_kernel<<<cdiv(rowTiles,8), b256, 0, stream>>>(G216, 128, 128, PkCLS, 8, out, 40, 40, clsB, rowTiles);
}
